// SwinTransformerBlock_12610023981396
// MI455X (gfx1250) — matrix operand order repaired and checked by reading, not compiled, not run on hardware
//
#include <hip/hip_runtime.h>
#include <hip/hip_bf16.h>
#include <math.h>

// ---------------- problem constants ----------------
#define TOK   200704      // B*H*W = 64*56*56
#define CH    192
#define NHEAD 6
#define HD    32
#define NTOK  49          // tokens per window
#define WINDOWS 4096      // B * 64
#define SHIFT 3
#define HH    56
#define ATT_SCALE 0.17677669529663687f   // 1/sqrt(32)

typedef __attribute__((ext_vector_type(16))) __bf16 v16bf;
typedef __attribute__((ext_vector_type(8)))  __bf16 v8bf;
typedef __attribute__((ext_vector_type(8)))  float  v8f;

// CDNA5 async memory->LDS path (ASYNCcnt tracked), guarded for toolchain drift.
#if defined(__HIP_DEVICE_COMPILE__) && \
    __has_builtin(__builtin_amdgcn_global_load_async_to_lds_b128) && \
    __has_builtin(__builtin_amdgcn_s_wait_asynccnt)
#define HAVE_ASYNC_LDS 1
#else
#define HAVE_ASYNC_LDS 0
#endif

#if HAVE_ASYNC_LDS
typedef int vi4 __attribute__((vector_size(16)));                 // int4, matches builtin pointee
typedef __attribute__((address_space(1))) vi4* gvi4p;             // global (device) int4*
typedef __attribute__((address_space(3))) vi4* lvi4p;             // LDS int4*

__device__ __forceinline__ void async_copy16(const void* g, void* l) {
  // GLOBAL_LOAD_ASYNC_TO_LDS_B128: per-lane 16B memory->LDS DMA, ASYNCcnt tracked
  __builtin_amdgcn_global_load_async_to_lds_b128(
      (gvi4p)(vi4*)g, (lvi4p)(vi4*)l, 0, 0);
}
#endif

__device__ __forceinline__ v8f wmma_bf16(v16bf a, v16bf b, v8f c) {
  // v_wmma_f32_16x16x32_bf16  D = A*B + C, f32 accumulate
  return __builtin_amdgcn_wmma_f32_16x16x32_bf16(false, a, false, b, (short)0, c, false, false);
}

// A-fragment (16x32 bf16, M x K) in contiguous k order (same as the B packing):
// lane l: row m = l%16; elements 0..15 -> K = (l/16)*16 + 0..15
// => two contiguous 16-byte loads from a row-major row.
__device__ __forceinline__ v16bf make_afrag(const __bf16* rowptr, int kbase, int g) {
  v8bf lo = *(const v8bf*)(rowptr + kbase + g * 16);
  v8bf hi = *(const v8bf*)(rowptr + kbase + g * 16 + 8);
  return __builtin_shufflevector(lo, hi, 0,1,2,3,4,5,6,7,8,9,10,11,12,13,14,15);
}

__device__ __forceinline__ int region9(int x) { return (x < 49) ? 0 : ((x < 53) ? 1 : 2); }

// ---------------- weight pre-pack into B-fragment order ----------------
// B-fragment (32x16 bf16, K x N): lane l holds n = l%16, K = (l/16)*16 + e (e=0..15)
// packed[((kb*NT+nt)*32 + lane)*16 + e] = W[kb*32 + (lane/16)*16 + e][nt*16 + lane%16]
__global__ void pack_w(const float* __restrict__ W, __bf16* __restrict__ Wp, int K, int Nc) {
  int NT = Nc >> 4;
  int total = K * Nc;
  for (int idx = blockIdx.x * blockDim.x + threadIdx.x; idx < total; idx += gridDim.x * blockDim.x) {
    int e = idx & 15;
    int l = (idx >> 4) & 31;
    int tile = idx >> 9;
    int nt = tile % NT;
    int kb = tile / NT;
    int k = kb * 32 + (l >> 4) * 16 + e;
    int n = nt * 16 + (l & 15);
    Wp[idx] = (__bf16)W[(size_t)k * Nc + n];
  }
}

// ---------------- LN1 fused with roll(-3,-3) + window partition, emits bf16 ----------------
__global__ void ln1_win(const float* __restrict__ x, const float* __restrict__ lw,
                        const float* __restrict__ lb, __bf16* __restrict__ xw) {
  int wid = threadIdx.x >> 5, lane = threadIdx.x & 31;
  int tok = blockIdx.x * 8 + wid;            // destination (windowed) token
  int wdx = tok / NTOK, n = tok % NTOK;
  int b  = wdx >> 6, nw = wdx & 63;
  int wh = nw >> 3,  ww = nw & 7;
  int i = n / 7, j = n % 7;
  int hs = wh * 7 + i + SHIFT; if (hs >= HH) hs -= HH;   // rolled source row
  int ws = ww * 7 + j + SHIFT; if (ws >= HH) ws -= HH;
  const float* src = x + ((size_t)b * 3136 + hs * 56 + ws) * CH;
  float v[6]; float s = 0.f, sq = 0.f;
#pragma unroll
  for (int t = 0; t < 6; t++) { float f = src[t * 32 + lane]; v[t] = f; s += f; sq += f * f; }
#pragma unroll
  for (int off = 16; off; off >>= 1) { s += __shfl_xor(s, off, 32); sq += __shfl_xor(sq, off, 32); }
  float mean = s * (1.f / 192.f);
  float var  = sq * (1.f / 192.f) - mean * mean;
  float rstd = rsqrtf(var + 1e-5f);
  __bf16* dst = xw + (size_t)tok * CH;
#pragma unroll
  for (int t = 0; t < 6; t++) {
    int c = t * 32 + lane;
    dst[c] = (__bf16)((v[t] - mean) * rstd * lw[c] + lb[c]);
  }
}

// ---------------- plain LN (LN2) -> bf16 ----------------
__global__ void ln_plain(const float* __restrict__ x, const float* __restrict__ lw,
                         const float* __restrict__ lb, __bf16* __restrict__ o) {
  int wid = threadIdx.x >> 5, lane = threadIdx.x & 31;
  int tok = blockIdx.x * 8 + wid;
  const float* src = x + (size_t)tok * CH;
  float v[6]; float s = 0.f, sq = 0.f;
#pragma unroll
  for (int t = 0; t < 6; t++) { float f = src[t * 32 + lane]; v[t] = f; s += f; sq += f * f; }
#pragma unroll
  for (int off = 16; off; off >>= 1) { s += __shfl_xor(s, off, 32); sq += __shfl_xor(sq, off, 32); }
  float mean = s * (1.f / 192.f);
  float var  = sq * (1.f / 192.f) - mean * mean;
  float rstd = rsqrtf(var + 1e-5f);
  __bf16* dst = o + (size_t)tok * CH;
#pragma unroll
  for (int t = 0; t < 6; t++) {
    int c = t * 32 + lane;
    dst[c] = (__bf16)((v[t] - mean) * rstd * lw[c] + lb[c]);
  }
}

// ---------------- QKV GEMM: [TOK,192] x [192,576] + bias -> bf16 ----------------
__global__ void gemm_qkv(const __bf16* __restrict__ A, const __bf16* __restrict__ Wp,
                         const float* __restrict__ bias, __bf16* __restrict__ Out) {
  const int K = CH, Nc = 576, NT = 36;
  int wid = threadIdx.x >> 5, lane = threadIdx.x & 31, g = lane >> 4;
  int mt = wid & 3, nh = wid >> 2;
  int row0 = blockIdx.x * 64 + mt * 16;
  int ntb  = blockIdx.y * 4 + nh * 2;
  const __bf16* arow = A + (size_t)(row0 + (lane & 15)) * K;
  v8f acc0 = {}; v8f acc1 = {};
#pragma unroll
  for (int kb = 0; kb < 6; kb++) {
    v16bf a  = make_afrag(arow, kb * 32, g);
    v16bf b0 = *(const v16bf*)(Wp + ((size_t)(kb * NT + ntb)) * 512 + lane * 16);
    v16bf b1 = *(const v16bf*)(Wp + ((size_t)(kb * NT + ntb + 1)) * 512 + lane * 16);
    acc0 = wmma_bf16(a, b0, acc0);
    acc1 = wmma_bf16(a, b1, acc1);
  }
  int col0 = ntb * 16 + (lane & 15);
  float bb0 = bias[col0], bb1 = bias[col0 + 16];
#pragma unroll
  for (int r = 0; r < 8; r++) {
    int m = row0 + r + g * 8;
    Out[(size_t)m * Nc + col0]      = (__bf16)(acc0[r] + bb0);
    Out[(size_t)m * Nc + col0 + 16] = (__bf16)(acc1[r] + bb1);
  }
}

// ---------------- windowed attention, one workgroup per (window, head) ----------------
__global__ void attn_kernel(const __bf16* __restrict__ qkv, const float* __restrict__ relb,
                            __bf16* __restrict__ outb) {
  int w = blockIdx.x, head = blockIdx.y;
  int tid = threadIdx.x;
  int wid = tid >> 5, lane = tid & 31, g = lane >> 4;
  __shared__ __align__(32) __bf16 qs[64 * 32];
  __shared__ __align__(32) __bf16 ksm[64 * 32];
  __shared__ __align__(32) __bf16 vT[32 * 64];
  __shared__ float sS[64 * 64];
  __shared__ __align__(32) __bf16 pS[64 * 64];

  const __bf16 z = (__bf16)0.f;
  size_t base = (size_t)w * NTOK * 576 + head * 32;

#if HAVE_ASYNC_LDS
  // zero the pad rows (tokens 49..63) with DS stores (drained by the barrier)
  for (int idx = tid; idx < 15 * 32; idx += 128) {
    int t = 49 + (idx >> 5), d = idx & 31;
    qs[t * 32 + d] = z; ksm[t * 32 + d] = z; vT[d * 64 + t] = z;
  }
  // q/k rows stream memory->LDS via the async DMA path (ASYNCcnt),
  // 16B chunks, per-lane global+LDS addresses. 49 rows * 4 chunks = 196.
  for (int c = tid; c < 196; c += 128) {
    int t = c >> 2, p = c & 3;
    const __bf16* gq = qkv + base + (size_t)t * 576 + p * 8;
    async_copy16(gq,       qs  + t * 32 + p * 8);
    async_copy16(gq + 192, ksm + t * 32 + p * 8);
  }
  // v needs a transpose into vT -> VGPR path
  for (int idx = tid; idx < NTOK * 32; idx += 128) {
    int t = idx >> 5, d = idx & 31;
    vT[d * 64 + t] = qkv[base + (size_t)t * 576 + 384 + d];
  }
  __builtin_amdgcn_s_wait_asynccnt(0);
  __syncthreads();
#else
  for (int idx = tid; idx < 2048; idx += 128) {
    int t = idx >> 5, d = idx & 31;
    __bf16 qv = z, kv = z, vv = z;
    if (t < NTOK) {
      const __bf16* p = qkv + base + (size_t)t * 576 + d;
      qv = p[0]; kv = p[192]; vv = p[384];
    }
    qs[idx] = qv; ksm[idx] = kv; vT[d * 64 + t] = vv;
  }
  __syncthreads();
#endif

  // S = q @ k^T  (64x64 padded, K=32)
  const __bf16* arow = qs + (wid * 16 + (lane & 15)) * 32;
  v16bf aq = make_afrag(arow, 0, g);
  int nw = w & 63; int whh = nw >> 3, www = nw & 7;
#pragma unroll
  for (int nt = 0; nt < 4; nt++) {
    v16bf bk = *(const v16bf*)(ksm + (nt * 16 + (lane & 15)) * 32 + g * 16);
    v8f acc = {};
    acc = wmma_bf16(aq, bk, acc);
    int n = nt * 16 + (lane & 15);
#pragma unroll
    for (int r = 0; r < 8; r++) {
      int m = wid * 16 + r + g * 8;
      float val = -1e30f;
      if (m < NTOK && n < NTOK) {
        int i1 = m / 7, j1 = m % 7, i2 = n / 7, j2 = n % 7;
        float bias = relb[((i1 - i2 + 6) * 13 + (j1 - j2 + 6)) * NHEAD + head];
        int r1 = region9(whh * 7 + i1) * 3 + region9(www * 7 + j1);
        int r2 = region9(whh * 7 + i2) * 3 + region9(www * 7 + j2);
        val = acc[r] * ATT_SCALE + bias + ((r1 != r2) ? -100.f : 0.f);
      }
      sS[m * 64 + n] = val;
    }
  }
  __syncthreads();

  // row softmax (rows 0..63; padded cols underflow to 0)
  if (tid < 64) {
    float mx = -1e30f;
    for (int n2 = 0; n2 < 64; n2++) mx = fmaxf(mx, sS[tid * 64 + n2]);
    float sum = 0.f;
    for (int n2 = 0; n2 < 64; n2++) sum += expf(sS[tid * 64 + n2] - mx);
    float inv = 1.f / sum;
    for (int n2 = 0; n2 < 64; n2++) pS[tid * 64 + n2] = (__bf16)(expf(sS[tid * 64 + n2] - mx) * inv);
  }
  __syncthreads();

  // O = P @ V   (64x32, K=64)
  const __bf16* prow = pS + (wid * 16 + (lane & 15)) * 64;
#pragma unroll
  for (int nt = 0; nt < 2; nt++) {
    v8f acc = {};
#pragma unroll
    for (int kb = 0; kb < 2; kb++) {
      v16bf ap = make_afrag(prow, kb * 32, g);
      v16bf bv = *(const v16bf*)(vT + (nt * 16 + (lane & 15)) * 64 + kb * 32 + g * 16);
      acc = wmma_bf16(ap, bv, acc);
    }
    int d = nt * 16 + (lane & 15);
#pragma unroll
    for (int r = 0; r < 8; r++) {
      int m = wid * 16 + r + g * 8;
      if (m < NTOK)
        outb[((size_t)w * NTOK + m) * CH + head * 32 + d] = (__bf16)acc[r];
    }
  }
}

// ---------------- proj GEMM + window-reverse + roll(+3,+3) + residual -> x1 (f32) ----------------
__global__ void gemm_proj(const __bf16* __restrict__ A, const __bf16* __restrict__ Wp,
                          const float* __restrict__ bias, const float* __restrict__ xin,
                          float* __restrict__ x1) {
  const int K = CH, NT = 12;
  int wid = threadIdx.x >> 5, lane = threadIdx.x & 31, g = lane >> 4;
  int mt = wid & 3, nh = wid >> 2;
  int row0 = blockIdx.x * 64 + mt * 16;
  int ntb  = blockIdx.y * 4 + nh * 2;
  const __bf16* arow = A + (size_t)(row0 + (lane & 15)) * K;
  v8f acc0 = {}; v8f acc1 = {};
#pragma unroll
  for (int kb = 0; kb < 6; kb++) {
    v16bf a  = make_afrag(arow, kb * 32, g);
    v16bf b0 = *(const v16bf*)(Wp + ((size_t)(kb * NT + ntb)) * 512 + lane * 16);
    v16bf b1 = *(const v16bf*)(Wp + ((size_t)(kb * NT + ntb + 1)) * 512 + lane * 16);
    acc0 = wmma_bf16(a, b0, acc0);
    acc1 = wmma_bf16(a, b1, acc1);
  }
  int col0 = ntb * 16 + (lane & 15);
  float bb0 = bias[col0], bb1 = bias[col0 + 16];
#pragma unroll
  for (int r = 0; r < 8; r++) {
    int m = row0 + r + g * 8;                 // windowed token
    int wdx = m / NTOK, n = m % NTOK;
    int b  = wdx >> 6, nw = wdx & 63;
    int wh = nw >> 3,  ww = nw & 7;
    int i = n / 7, j = n % 7;
    int ho = wh * 7 + i + SHIFT; if (ho >= HH) ho -= HH;
    int wo = ww * 7 + j + SHIFT; if (wo >= HH) wo -= HH;
    size_t dst = ((size_t)b * 3136 + ho * 56 + wo) * CH;
    x1[dst + col0]      = xin[dst + col0]      + acc0[r] + bb0;
    x1[dst + col0 + 16] = xin[dst + col0 + 16] + acc1[r] + bb1;
  }
}

// ---------------- fused MLP: gelu(h2 @ W1 + b1) @ W2 + b2 + x1 -> out ----------------
// hidden 64x768 bf16 tile lives in LDS (96 KB of the 320 KB WGP LDS)
__global__ void mlp_fused(const __bf16* __restrict__ h2, const __bf16* __restrict__ W1p,
                          const float* __restrict__ b1, const __bf16* __restrict__ W2p,
                          const float* __restrict__ b2, const float* __restrict__ x1,
                          float* __restrict__ out) {
  __shared__ __align__(32) __bf16 hid[64 * 768];
  int tid = threadIdx.x, wid = tid >> 5, lane = tid & 31, g = lane >> 4;
  int mt = wid & 3;
  size_t row0 = (size_t)blockIdx.x * 64 + mt * 16;

  // phase 1: hidden = gelu(h2 @ W1 + b1)
  const __bf16* arow = h2 + (row0 + (lane & 15)) * CH;
  v16bf afr[6];
#pragma unroll
  for (int kb = 0; kb < 6; kb++) afr[kb] = make_afrag(arow, kb * 32, g);
  for (int nt = (wid >> 2); nt < 48; nt += 2) {
    v8f acc = {};
#pragma unroll
    for (int kb = 0; kb < 6; kb++) {
      v16bf b = *(const v16bf*)(W1p + ((size_t)(kb * 48 + nt)) * 512 + lane * 16);
      acc = wmma_bf16(afr[kb], b, acc);
    }
    int col = nt * 16 + (lane & 15);
    float bb = b1[col];
#pragma unroll
    for (int r = 0; r < 8; r++) {
      int m = mt * 16 + r + g * 8;
      float xv = acc[r] + bb;
      hid[m * 768 + col] = (__bf16)(0.5f * xv * (1.f + erff(xv * 0.70710678118654752f)));
    }
  }
  __syncthreads();

  // phase 2: out = x1 + hidden @ W2 + b2
  const __bf16* hrow = hid + (mt * 16 + (lane & 15)) * 768;
  for (int nt = (wid >> 2); nt < 12; nt += 2) {
    v8f acc = {};
#pragma unroll
    for (int kb = 0; kb < 24; kb++) {
      v16bf a = make_afrag(hrow, kb * 32, g);
      v16bf b = *(const v16bf*)(W2p + ((size_t)(kb * 12 + nt)) * 512 + lane * 16);
      acc = wmma_bf16(a, b, acc);
    }
    int col = nt * 16 + (lane & 15);
    float bb = b2[col];
#pragma unroll
    for (int r = 0; r < 8; r++) {
      size_t tok = row0 + r + g * 8;
      out[tok * CH + col] = x1[tok * CH + col] + acc[r] + bb;
    }
  }
}

// ---------------- launcher ----------------
extern "C" void kernel_launch(void* const* d_in, const int* in_sizes, int n_in,
                              void* d_out, int out_size, void* d_ws, size_t ws_size,
                              hipStream_t stream) {
  (void)in_sizes; (void)n_in; (void)out_size; (void)ws_size;
  const float* x     = (const float*)d_in[0];
  const float* ln1w  = (const float*)d_in[1];
  const float* ln1b  = (const float*)d_in[2];
  const float* qkvw  = (const float*)d_in[3];
  const float* qkvb  = (const float*)d_in[4];
  const float* projw = (const float*)d_in[5];
  const float* projb = (const float*)d_in[6];
  const float* relb  = (const float*)d_in[7];
  const float* ln2w  = (const float*)d_in[8];
  const float* ln2b  = (const float*)d_in[9];
  const float* fc1w  = (const float*)d_in[10];
  const float* fc1b  = (const float*)d_in[11];
  const float* fc2w  = (const float*)d_in[12];
  const float* fc2b  = (const float*)d_in[13];
  float* out = (float*)d_out;

  char* ws = (char*)d_ws;
  size_t o = 0;
  __bf16* xw   = (__bf16*)(ws + o); o += (size_t)TOK * CH * 2;     // 77 MB (later reused as h2)
  __bf16* qkv  = (__bf16*)(ws + o); o += (size_t)TOK * 576 * 2;    // 231 MB
  __bf16* attn = (__bf16*)(ws + o); o += (size_t)TOK * CH * 2;     // 77 MB
  float*  x1   = (float*) (ws + o); o += (size_t)TOK * CH * 4;     // 154 MB
  __bf16* wq   = (__bf16*)(ws + o); o += (size_t)192 * 576 * 2;
  __bf16* wp   = (__bf16*)(ws + o); o += (size_t)192 * 192 * 2;
  __bf16* w1   = (__bf16*)(ws + o); o += (size_t)192 * 768 * 2;
  __bf16* w2   = (__bf16*)(ws + o); o += (size_t)768 * 192 * 2;
  __bf16* h2   = xw;  // reuse: xw dead after QKV GEMM

  pack_w<<<256, 256, 0, stream>>>(qkvw,  wq, 192, 576);
  pack_w<<<256, 256, 0, stream>>>(projw, wp, 192, 192);
  pack_w<<<256, 256, 0, stream>>>(fc1w,  w1, 192, 768);
  pack_w<<<256, 256, 0, stream>>>(fc2w,  w2, 768, 192);

  ln1_win<<<TOK / 8, 256, 0, stream>>>(x, ln1w, ln1b, xw);
  gemm_qkv<<<dim3(TOK / 64, 9), 256, 0, stream>>>(xw, wq, qkvb, qkv);
  attn_kernel<<<dim3(WINDOWS, NHEAD), 128, 0, stream>>>(qkv, relb, attn);
  gemm_proj<<<dim3(TOK / 64, 3), 256, 0, stream>>>(attn, wp, projb, x, x1);
  ln_plain<<<TOK / 8, 256, 0, stream>>>(x1, ln2w, ln2b, h2);
  mlp_fused<<<TOK / 64, 256, 0, stream>>>(h2, w1, fc1b, w2, fc2b, x1, out);
}
